// DynaMicConv_47536698032558
// MI455X (gfx1250) — matrix operand order repaired and checked by reading, not compiled, not run on hardware
//
#include <hip/hip_runtime.h>

#define G_    14
#define P_    196
#define COUT_ 768
#define K_    768
#define B_    64
#define NT_   128   // N tile per workgroup
#define KC_   128   // K chunk of A staged in LDS (f32)
#define KS_   32    // K per WMMA step
#define AP_   132   // padded A row stride (floats) -> conflict-free ds reads
#define BP_   36    // padded B row stride (floats)

typedef __attribute__((ext_vector_type(16))) __bf16 v16bf;
typedef __attribute__((ext_vector_type(8)))  float  v8f;
typedef __attribute__((ext_vector_type(4)))  float  f4;

// Vector-of-4-int pointer types matching the async-LDS builtin's signature
// (param type reported by clang: "int __vector(4) __device__ *").
typedef int v4i_ __attribute__((vector_size(16)));
typedef __attribute__((address_space(1))) v4i_ gv4i;   // global (AS1)
typedef __attribute__((address_space(3))) v4i_ lv4i;   // LDS (AS3)

#if defined(__gfx1250__) && __has_builtin(__builtin_amdgcn_global_load_async_to_lds_b128)
#define ASYNC_LDS 1
#else
#define ASYNC_LDS 0
#endif

__device__ __forceinline__ void copy16(const float* g, float* l) {
#if ASYNC_LDS
  __builtin_amdgcn_global_load_async_to_lds_b128((gv4i*)g, (lv4i*)l, 0, 0);
#else
  *(f4*)l = *(const f4*)g;
#endif
}

__device__ __forceinline__ void wait_async() {
#if ASYNC_LDS
#if __has_builtin(__builtin_amdgcn_s_wait_asynccnt)
  __builtin_amdgcn_s_wait_asynccnt(0);
#else
  asm volatile("s_wait_asynccnt 0" ::: "memory");
#endif
#endif
}

// A-matrix 16x32 bf16 fragment (ISA 7.12.2): lane row M=l&15;
// low lanes hold K=c0..c0+7 and c0+16..c0+23 with c0=0; high lanes c0=8.
__device__ __forceinline__ v16bf fragA(const float* base, int c0) {
  f4 a0 = *(const f4*)(base + c0);
  f4 a1 = *(const f4*)(base + c0 + 4);
  f4 b0 = *(const f4*)(base + c0 + 16);
  f4 b1 = *(const f4*)(base + c0 + 20);
  v16bf f;
  f[0]=(__bf16)a0.x; f[1]=(__bf16)a0.y; f[2]=(__bf16)a0.z; f[3]=(__bf16)a0.w;
  f[4]=(__bf16)a1.x; f[5]=(__bf16)a1.y; f[6]=(__bf16)a1.z; f[7]=(__bf16)a1.w;
  f[8]=(__bf16)b0.x; f[9]=(__bf16)b0.y; f[10]=(__bf16)b0.z; f[11]=(__bf16)b0.w;
  f[12]=(__bf16)b1.x; f[13]=(__bf16)b1.y; f[14]=(__bf16)b1.z; f[15]=(__bf16)b1.w;
  return f;
}

// B-matrix 32x16 bf16 fragment: lane col N=l&15; low lanes hold
// K=c0..c0+7 and c0+16..c0+23 with c0=0; high lanes c0=8.
__device__ __forceinline__ v16bf fragB(const float* base, int c0) {
  f4 a0 = *(const f4*)(base + c0);
  f4 a1 = *(const f4*)(base + c0 + 4);
  f4 a2 = *(const f4*)(base + c0 + 16);
  f4 a3 = *(const f4*)(base + c0 + 20);
  v16bf f;
  f[0]=(__bf16)a0.x; f[1]=(__bf16)a0.y; f[2]=(__bf16)a0.z; f[3]=(__bf16)a0.w;
  f[4]=(__bf16)a1.x; f[5]=(__bf16)a1.y; f[6]=(__bf16)a1.z; f[7]=(__bf16)a1.w;
  f[8]=(__bf16)a2.x; f[9]=(__bf16)a2.y; f[10]=(__bf16)a2.z; f[11]=(__bf16)a2.w;
  f[12]=(__bf16)a3.x; f[13]=(__bf16)a3.y; f[14]=(__bf16)a3.z; f[15]=(__bf16)a3.w;
  return f;
}

__global__ __launch_bounds__(256)
void patch_gemm_wmma(const float* __restrict__ x, const float* __restrict__ W,
                     const float* __restrict__ bias, float* __restrict__ out) {
  __shared__ float Ach[B_ * AP_];   // 64 x (128+4) f32  = 33.0 KB
  __shared__ float Btl[NT_ * BP_];  // 128 x (32+4) f32  = 18.0 KB

  const int t       = threadIdx.x;
  const int lane    = t & 31;
  const int wave    = t >> 5;
  const int m_block = wave & 3;     // 4 M-subtiles of 16
  const int n_group = wave >> 2;    // 2 N-groups of 64
  const int hiA     = (lane & 16) ? 8 : 0;
  const int hiB     = (lane & 16) ? 8 : 0;

  const int p     = blockIdx.x / (COUT_ / NT_);
  const int nt    = blockIdx.x % (COUT_ / NT_);
  const int nbase = nt * NT_;
  const int gi    = p / G_;
  const int gj    = p % G_;

  v8f acc[4] = {};

  for (int cc = 0; cc < K_ / KC_; ++cc) {        // 6 K-chunks of 128
    const int ch  = cc >> 1;                     // input channel
    const int pyb = (cc & 1) * 8;                // half-channel row block
    for (int kt = 0; kt < KC_ / KS_; ++kt) {     // 4 WMMA K-steps per chunk
      __syncthreads();                           // prior round's LDS reads done
      if (kt == 0) {
        // Stage A chunk: 64 rows x 128 K (f32). 2048 16B segments, 8/thread.
        #pragma unroll
        for (int i = 0; i < 8; ++i) {
          int s   = t + i * 256;
          int seg = s & 3;
          int run = s >> 2;          // run = m*8 + py (16 contiguous floats)
          int py  = run & 7;
          int m   = run >> 3;
          const float* g = x + ((size_t)(m * 3 + ch) * 224 + gi * 16 + pyb + py) * 224
                             + gj * 16 + seg * 4;
          copy16(g, &Ach[m * AP_ + py * 16 + seg * 4]);
        }
      }
      const int k0 = cc * KC_ + kt * KS_;
      // Stage B tile: 128 n-rows x 32 K (f32). 1024 16B segments, 4/thread.
      #pragma unroll
      for (int i = 0; i < 4; ++i) {
        int s  = t + i * 256;
        int ks = s & 7;
        int n  = s >> 3;
        const float* g = W + ((size_t)p * COUT_ + nbase + n) * K_ + k0 + ks * 4;
        copy16(g, &Btl[n * BP_ + ks * 4]);
      }
      if (k0 + KS_ < K_) {  // prefetch next W tile into GL2
        __builtin_prefetch(W + ((size_t)p * COUT_ + nbase + (t >> 1)) * K_ + k0 + KS_, 0, 0);
      }
      wait_async();
      __syncthreads();

      const int mrow = m_block * 16 + (lane & 15);
      v16bf a = fragA(&Ach[mrow * AP_ + kt * KS_], hiA);
      #pragma unroll
      for (int ns = 0; ns < 4; ++ns) {
        const int nrow = n_group * 64 + ns * 16 + (lane & 15);
        v16bf b = fragB(&Btl[nrow * BP_], hiB);
        acc[ns] = __builtin_amdgcn_wmma_f32_16x16x32_bf16(
            false, a, false, b, (short)0, acc[ns], false, false);
      }
    }
  }

  // Epilogue: D layout (ISA 7.12.2): VGPR r holds M=r (lanes<16) / M=r+8.
  #pragma unroll
  for (int ns = 0; ns < 4; ++ns) {
    const int o  = nbase + n_group * 64 + ns * 16 + (lane & 15);
    const float bv = bias[p * COUT_ + o];
    #pragma unroll
    for (int r = 0; r < 8; ++r) {
      const int m = m_block * 16 + hiA + r;
      out[((size_t)m * COUT_ + o) * P_ + p] = acc[ns][r] + bv;
    }
  }
}

extern "C" void kernel_launch(void* const* d_in, const int* in_sizes, int n_in,
                              void* d_out, int out_size, void* d_ws, size_t ws_size,
                              hipStream_t stream) {
  (void)in_sizes; (void)n_in; (void)d_ws; (void)ws_size; (void)out_size;
  const float* x    = (const float*)d_in[0];
  const float* W    = (const float*)d_in[1];
  const float* bias = (const float*)d_in[2];
  float* out        = (float*)d_out;

  dim3 grid(P_ * (COUT_ / NT_));   // 196 * 6 = 1176 workgroups
  dim3 block(256);                 // 8 wave32 waves
  patch_gemm_wmma<<<grid, block, 0, stream>>>(x, W, bias, out);
}
